// MultiheadAttention_9758165696939
// MI455X (gfx1250) — matrix operand order repaired and checked by reading, not compiled, not run on hardware
//
#include <hip/hip_runtime.h>
#include <hip/hip_bf16.h>

typedef __attribute__((ext_vector_type(16))) _Float16 v16h;
typedef __attribute__((ext_vector_type(8)))  _Float16 v8h;
typedef __attribute__((ext_vector_type(8)))  float    v8f;

#define D_MODEL 256
#define NSEQ    4096
#define NBATCH  2
#define NHEADS  8
#define DH      32

// softmax_scale * log2(e), folded into Q at projection time so the
// attention inner loop works directly in the base-2 domain.
#define Q_PRESCALE 0.2550349086f

#if defined(__HIP_DEVICE_COMPILE__) && defined(__has_builtin)
#if __has_builtin(__builtin_amdgcn_global_load_async_to_lds_b128) && \
    __has_builtin(__builtin_amdgcn_s_wait_asynccnt)
#define USE_ASYNC 1
#endif
#if __has_builtin(__builtin_amdgcn_update_dpp)
#define USE_DPP 1
#endif
#if __has_builtin(__builtin_amdgcn_exp2f)
#define USE_RAW_EXP2 1
#endif
#endif
#ifndef USE_ASYNC
#define USE_ASYNC 0
#endif
#ifndef USE_DPP
#define USE_DPP 0
#endif
#ifndef USE_RAW_EXP2
#define USE_RAW_EXP2 0
#endif

#if USE_ASYNC
typedef int v4i_vs __attribute__((__vector_size__(16)));
typedef __attribute__((address_space(1))) v4i_vs* as1_v4i;
typedef __attribute__((address_space(3))) v4i_vs* as3_v4i;
#endif

union V16 { v16h v; v8h h[2]; };

__device__ __forceinline__ float fast_exp2(float x) {
#if USE_RAW_EXP2
    return __builtin_amdgcn_exp2f(x);   // raw v_exp_f32 (args <= 0 here)
#else
    return exp2f(x);
#endif
}

// Max-reduce across the 16 lanes of each wave32 half-row (pure VALU via DPP).
__device__ __forceinline__ float row16_max(float x) {
#if USE_DPP
    union { float f; int i; } a, b;
    a.f = x;
    b.i = __builtin_amdgcn_update_dpp(a.i, a.i, 0xB1,  0xF, 0xF, false); // quad_perm [1,0,3,2]
    a.f = fmaxf(a.f, b.f);
    b.i = __builtin_amdgcn_update_dpp(a.i, a.i, 0x4E,  0xF, 0xF, false); // quad_perm [2,3,0,1]
    a.f = fmaxf(a.f, b.f);
    b.i = __builtin_amdgcn_update_dpp(a.i, a.i, 0x141, 0xF, 0xF, false); // row_half_mirror
    a.f = fmaxf(a.f, b.f);
    b.i = __builtin_amdgcn_update_dpp(a.i, a.i, 0x140, 0xF, 0xF, false); // row_mirror
    a.f = fmaxf(a.f, b.f);
    return a.f;
#else
    x = fmaxf(x, __shfl_xor(x, 1));
    x = fmaxf(x, __shfl_xor(x, 2));
    x = fmaxf(x, __shfl_xor(x, 4));
    x = fmaxf(x, __shfl_xor(x, 8));
    return x;
#endif
}

// ---------------------------------------------------------------------------
// Flash attention: block = 8 waves = 128 query rows of one (b,h).
// Key tiles of 64 keys double-buffered in LDS via async global->LDS copies,
// shared by all 8 waves (one barrier per tile).
// Softmax row-sums are accumulated with an extra WMMA against an all-ones
// B matrix (denominator lives in a C-layout tile, replicated per lane).
// ---------------------------------------------------------------------------
__global__ __launch_bounds__(256) void flash_attn_kernel(
    const _Float16* __restrict__ Qh, const _Float16* __restrict__ Kh,
    const _Float16* __restrict__ Vt, float* __restrict__ out)
{
    __shared__ __align__(16) _Float16 Ks[2][64 * 32];    // [buf][key][dh]
    __shared__ __align__(16) _Float16 Vs[2][32 * 64];    // [buf][dh][key]
    __shared__ __align__(16) _Float16 Ps[8 * 16 * 64];   // per-wave P tiles

    const int tid  = threadIdx.x;
    const int lane = tid & 31;
    const int w    = tid >> 5;
    const int ml   = lane & 15;
    const int hi   = lane >> 4;

    const int h    = blockIdx.y;
    const int bidx = blockIdx.z;
    const int bh   = bidx * NHEADS + h;
    const int q0   = blockIdx.x * 128 + w * 16;

    const _Float16* Qbase = Qh + (size_t)bh * NSEQ * DH;
    const _Float16* Kbase = Kh + (size_t)bh * NSEQ * DH;
    const _Float16* Vbase = Vt + (size_t)bh * DH * NSEQ;

    // Issue cooperative copy of one 64-key K/V tile into LDS buffer `buf`.
    auto issue_tile = [&](int kt, int buf) {
        const uint4* ksrc = ((const uint4*)(Kbase + (size_t)kt * 64 * DH)) + tid;
        uint4*       kdst = ((uint4*)Ks[buf]) + tid;
        const int row = tid >> 3, chunk = tid & 7;
        const uint4* vsrc =
            (const uint4*)(Vbase + (size_t)row * NSEQ + kt * 64 + chunk * 8);
        uint4* vdst = (uint4*)(Vs[buf] + row * 64 + chunk * 8);
#if USE_ASYNC
        __builtin_amdgcn_global_load_async_to_lds_b128(
            (as1_v4i)ksrc, (as3_v4i)kdst, 0, 0);
        __builtin_amdgcn_global_load_async_to_lds_b128(
            (as1_v4i)vsrc, (as3_v4i)vdst, 0, 0);
#else
        *kdst = *ksrc;
        *vdst = *vsrc;
#endif
    };

    // Load Q A-tile (16x32), resident for whole kernel (pre-scaled in proj).
    V16 aq;
    {
        const _Float16* qp = Qbase + (size_t)(q0 + ml) * DH + hi * 16;
        aq.h[0] = *(const v8h*)qp;
        aq.h[1] = *(const v8h*)(qp + 8);
    }

    // All-ones B matrix for row-sum WMMAs.
    v16h ones;
#pragma unroll
    for (int i = 0; i < 16; ++i) ones[i] = (_Float16)1.0f;

    float ms[8];
#pragma unroll
    for (int j = 0; j < 8; ++j) ms[j] = -__builtin_inff();
    v8f acc0 = {}, acc1 = {}, lsacc = {};

    _Float16* Pw = Ps + w * (16 * 64);

    issue_tile(0, 0);   // prefetch first tile

    for (int kt = 0; kt < NSEQ / 64; ++kt) {
        const int cur = kt & 1;
#if USE_ASYNC
        __builtin_amdgcn_s_wait_asynccnt(0);
#endif
        __syncthreads();   // tile `cur` ready; previous compute on `cur^1` done
        if (kt + 1 < NSEQ / 64) issue_tile(kt + 1, cur ^ 1);

        const _Float16* Kcur = Ks[cur];
        const _Float16* Vcur = Vs[cur];

        // S = Q @ K^T : 4 wmma (16 q x 16 keys each), K-dim = dh = 32.
        // Q pre-scaled, so S is already a base-2 logit.
        v8f s[4];
#pragma unroll
        for (int g = 0; g < 4; ++g) {
            const _Float16* kp = Kcur + (g * 16 + ml) * 32 + hi * 16;
            V16 bk_;
            bk_.h[0] = *(const v8h*)kp;
            bk_.h[1] = *(const v8h*)(kp + 8);
            v8f z = {};
            s[g] = __builtin_amdgcn_wmma_f32_16x16x32_f16(false, aq.v, false, bk_.v,
                                                          (short)0, z, false, false);
        }

        // Online softmax (base-2). Row r = j + 8*hi lives on 16 lanes (N = ml).
#pragma unroll
        for (int j = 0; j < 8; ++j) {
            float mx = fmaxf(fmaxf(s[0][j], s[1][j]), fmaxf(s[2][j], s[3][j]));
            mx = row16_max(mx);
            const float mnew  = fmaxf(ms[j], mx);
            const float alpha = fast_exp2(ms[j] - mnew);
            ms[j] = mnew;
#pragma unroll
            for (int g = 0; g < 4; ++g) s[g][j] = fast_exp2(s[g][j] - mnew);
            acc0[j]  *= alpha;
            acc1[j]  *= alpha;
            lsacc[j] *= alpha;
        }

        // Stage P (C layout) to LDS so it can be re-read in A layout.
#pragma unroll
        for (int g = 0; g < 4; ++g)
#pragma unroll
            for (int j = 0; j < 8; ++j)
                Pw[(j + hi * 8) * 64 + g * 16 + ml] = (_Float16)s[g][j];

        // acc += P @ V ; lsacc += P @ ones  (two 32-key chunks)
#pragma unroll
        for (int h2 = 0; h2 < 2; ++h2) {
            const _Float16* pp = Pw + ml * 64 + h2 * 32 + hi * 16;
            V16 ap;
            ap.h[0] = *(const v8h*)pp;
            ap.h[1] = *(const v8h*)(pp + 8);

            const _Float16* vp0 = Vcur + ml * 64 + h2 * 32 + hi * 16;
            const _Float16* vp1 = Vcur + (16 + ml) * 64 + h2 * 32 + hi * 16;
            V16 b0, b1;
            b0.h[0] = *(const v8h*)vp0; b0.h[1] = *(const v8h*)(vp0 + 8);
            b1.h[0] = *(const v8h*)vp1; b1.h[1] = *(const v8h*)(vp1 + 8);

            acc0 = __builtin_amdgcn_wmma_f32_16x16x32_f16(false, ap.v, false, b0.v,
                                                          (short)0, acc0, false, false);
            acc1 = __builtin_amdgcn_wmma_f32_16x16x32_f16(false, ap.v, false, b1.v,
                                                          (short)0, acc1, false, false);
            lsacc = __builtin_amdgcn_wmma_f32_16x16x32_f16(false, ap.v, false, ones,
                                                           (short)0, lsacc, false, false);
        }
    }

    // Epilogue: normalize and write [B, N, 256] f32.
#pragma unroll
    for (int j = 0; j < 8; ++j) {
        const float inv = 1.0f / lsacc[j];
        const int q = q0 + j + 8 * hi;
        float* op = out + ((size_t)bidx * NSEQ + q) * D_MODEL + h * DH;
        op[ml]      = acc0[j] * inv;
        op[16 + ml] = acc1[j] * inv;
    }
}

// ---------------------------------------------------------------------------
// Projection: Y = X @ W^T + b for Q,K,V; stores f16 into workspace.
// Q,K layout: [B*H, NSEQ, 32]. V layout (transposed): [B*H, 32, NSEQ].
// Q is pre-scaled by softmax_scale*log2(e).
// ---------------------------------------------------------------------------
__global__ __launch_bounds__(256) void proj_qkv_kernel(
    const float* __restrict__ xq, const float* __restrict__ xkv,
    const float* __restrict__ Wq, const float* __restrict__ bq,
    const float* __restrict__ Wk, const float* __restrict__ bk,
    const float* __restrict__ Wv, const float* __restrict__ bv,
    _Float16* __restrict__ Qh, _Float16* __restrict__ Kh,
    _Float16* __restrict__ Vt)
{
    const int lane = threadIdx.x & 31;
    const int w    = threadIdx.x >> 5;
    const int tlin = blockIdx.y * 8 + w;     // 0..47
    const int mat  = tlin >> 4;              // 0=Q 1=K 2=V
    const int ntile = tlin & 15;             // 16 col tiles
    const int mtile = blockIdx.x;            // 512 row tiles (B*N/16)

    const float* X    = (mat == 0) ? xq : xkv;
    const float* W    = (mat == 0) ? Wq : (mat == 1) ? Wk : Wv;
    const float* bias = (mat == 0) ? bq : (mat == 1) ? bk : bv;

    const int ml = lane & 15;
    const int hi = lane >> 4;
    const int arow = mtile * 16 + ml;   // row of X  (M index via A layout)
    const int brow = ntile * 16 + ml;   // row of W  (= output column, B layout)

    v8f c = {};
#pragma unroll
    for (int kk = 0; kk < 8; ++kk) {
        // A tile 16x32: lane ml = M row; 16 contiguous K halfs starting at hi*16
        const int kbA = kk * 32 + hi * 16;
        const float* ap = X + (size_t)arow * D_MODEL + kbA;
        v16h a{};
#pragma unroll
        for (int i = 0; i < 8; ++i) {
            a[i]     = (_Float16)ap[i];
            a[8 + i] = (_Float16)ap[8 + i];
        }
        // B tile 32x16: lane ml = N col; 16 contiguous K halfs starting at hi*16
        const int kbB = kk * 32 + hi * 16;
        const float* bp = W + (size_t)brow * D_MODEL + kbB;
        v16h b{};
#pragma unroll
        for (int i = 0; i < 16; ++i) b[i] = (_Float16)bp[i];

        c = __builtin_amdgcn_wmma_f32_16x16x32_f16(false, a, false, b,
                                                   (short)0, c, false, false);
    }

    const float bn = bias[brow];
    const int col  = brow;            // C layout: N = lane&15 -> same as brow
    const int h    = col >> 5;
    const int dcol = col & 31;
#pragma unroll
    for (int j = 0; j < 8; ++j) {
        const int m    = mtile * 16 + j + hi * 8;  // C layout: M = j + 8*hi
        const int bidx = m >> 12;                  // / NSEQ
        const int seq  = m & (NSEQ - 1);
        const int bh   = bidx * NHEADS + h;
        const float val = c[j] + bn;
        if (mat == 0)
            Qh[((size_t)bh * NSEQ + seq) * DH + dcol] = (_Float16)(val * Q_PRESCALE);
        else if (mat == 1)
            Kh[((size_t)bh * NSEQ + seq) * DH + dcol] = (_Float16)val;
        else
            Vt[((size_t)bh * DH + dcol) * NSEQ + seq] = (_Float16)val;
    }
}

// ---------------------------------------------------------------------------
extern "C" void kernel_launch(void* const* d_in, const int* in_sizes, int n_in,
                              void* d_out, int out_size, void* d_ws, size_t ws_size,
                              hipStream_t stream) {
    (void)in_sizes; (void)n_in; (void)out_size; (void)ws_size;
    const float* xq = (const float*)d_in[0];
    const float* xkv = (const float*)d_in[1];
    const float* Wq = (const float*)d_in[2];
    const float* bq = (const float*)d_in[3];
    const float* Wk = (const float*)d_in[4];
    const float* bk = (const float*)d_in[5];
    const float* Wv = (const float*)d_in[6];
    const float* bv = (const float*)d_in[7];

    const size_t elems = (size_t)NBATCH * NHEADS * NSEQ * DH; // 2,097,152
    _Float16* Qh = (_Float16*)d_ws;
    _Float16* Kh = Qh + elems;
    _Float16* Vt = Kh + elems;

    // 512 row tiles x 48 (mat,ntile) tiles / 8 waves per block
    proj_qkv_kernel<<<dim3(512, 6), 256, 0, stream>>>(
        xq, xkv, Wq, bq, Wk, bk, Wv, bv, Qh, Kh, Vt);

    // 32 q-blocks x 8 heads x 2 batches
    flash_attn_kernel<<<dim3(NSEQ / 128, NHEADS, NBATCH), 256, 0, stream>>>(
        Qh, Kh, Vt, (float*)d_out);
}
